// TripletLoss_8040178778595
// MI455X (gfx1250) — compile-verified, matrix operand order checked by reading, not run on hardware
//
#include <hip/hip_runtime.h>

// ---------------------------------------------------------------------------
// Triplet loss w/ batch-hard mining on MI455X (gfx1250).
//   X (8192x512 f32) -> row-normalize -> G = Xn @ Xn^T via V_WMMA_F32_16X16X32_BF16
//   with 2-term bf16 error compensation (hi*hi + hi*lo + lo*hi ~ fp32 accuracy).
//   Fused masked row min/max epilogue (never materialize 256MB Gram matrix).
//   K-chunks streamed into double-buffered LDS with GLOBAL_LOAD_ASYNC_TO_LDS_B128
//   (ASYNCcnt), overlapping the next chunk's DMA with the current chunk's WMMAs.
//   Deterministic: float min/max via monotonic u32 encoding + atomicMin/Max.
// ---------------------------------------------------------------------------

#define NROWS   8192
#define DDIM    512
#define MARGINF 0.5f
#define BM      64          // rows per block
#define BN      128         // cols per block
#define KC      32          // K chunk (= one bf16 WMMA K)
#define NCHUNK  (DDIM / KC)
#define LDT     40          // padded LDS row stride (bf16 elems) -> conflict-free b128 frag loads
#define FLTMAXF 3.402823466e+38f

// byte offsets inside one LDS buffer
#define OFF_AHI 0
#define OFF_ALO (BM * LDT * 2)                       // 5120
#define OFF_BHI (2 * BM * LDT * 2)                   // 10240
#define OFF_BLO (2 * BM * LDT * 2 + BN * LDT * 2)    // 20480
#define BUFBYTES (2 * BM * LDT * 2 + 2 * BN * LDT * 2) // 30720 (x2 buffers = 60KB LDS)

typedef __attribute__((ext_vector_type(8)))  float  v8f;
typedef __attribute__((ext_vector_type(16))) __bf16 v16bf;
typedef __attribute__((ext_vector_type(8)))  __bf16 v8bf;

union Frag { v16bf v; v8bf h[2]; };

// float -> bf16 (round-to-nearest-even), bit-exact, no __bf16 scalar math needed
__device__ __forceinline__ unsigned short f2bf(float f) {
  unsigned u = __float_as_uint(f);
  unsigned r = 0x7FFFu + ((u >> 16) & 1u);
  return (unsigned short)((u + r) >> 16);
}
__device__ __forceinline__ float bf2f(unsigned short h) {
  return __uint_as_float(((unsigned)h) << 16);
}
// monotonic float<->u32 map: f1<f2  <=>  encf(f1)<encf(f2)  (atomic min/max exact+deterministic)
__device__ __forceinline__ unsigned encf(float f) {
  unsigned u = __float_as_uint(f);
  return (u & 0x80000000u) ? ~u : (u | 0x80000000u);
}
__device__ __forceinline__ float decf(unsigned u) {
  return (u & 0x80000000u) ? __uint_as_float(u ^ 0x80000000u) : __uint_as_float(~u);
}

// ---- gfx1250 async memory->LDS DMA (tracked by ASYNCcnt, in-order completion) ----
__device__ __forceinline__ void async_ld_b128(unsigned lds_off, const unsigned short* sbase,
                                              unsigned voff_bytes) {
  asm volatile("global_load_async_to_lds_b128 %0, %1, %2"
               :: "v"(lds_off), "v"(voff_bytes), "s"(sbase)
               : "memory");
}
__device__ __forceinline__ void wait_async_le6() {
#if __has_builtin(__builtin_amdgcn_s_wait_asynccnt)
  __builtin_amdgcn_s_wait_asynccnt(6);
#else
  asm volatile("s_wait_asynccnt 0x6" ::: "memory");
#endif
}
__device__ __forceinline__ void wait_async_0() {
#if __has_builtin(__builtin_amdgcn_s_wait_asynccnt)
  __builtin_amdgcn_s_wait_asynccnt(0);
#else
  asm volatile("s_wait_asynccnt 0x0" ::: "memory");
#endif
}

// ---------------------------------------------------------------------------
// Kernel 1: per-row L2 normalize, split into bf16 hi/lo, init row sentinels.
// One wave (32 lanes) per row, 8 rows per 256-thread block.
// ---------------------------------------------------------------------------
__global__ __launch_bounds__(256) void norm_split_kernel(
    const float* __restrict__ X,
    unsigned short* __restrict__ Xhi, unsigned short* __restrict__ Xlo,
    unsigned* __restrict__ minpos, unsigned* __restrict__ maxneg)
{
  const int row  = blockIdx.x * 8 + (threadIdx.x >> 5);
  const int lane = threadIdx.x & 31;
  const float* xr = X + (size_t)row * DDIM;

  float v[16];
  float ss = 0.f;
#pragma unroll
  for (int i = 0; i < 4; ++i) {
    const float4 t = *(const float4*)(xr + (lane + 32 * i) * 4);
    v[4*i+0] = t.x; v[4*i+1] = t.y; v[4*i+2] = t.z; v[4*i+3] = t.w;
    ss += t.x*t.x + t.y*t.y + t.z*t.z + t.w*t.w;
  }
#pragma unroll
  for (int off = 16; off >= 1; off >>= 1) ss += __shfl_xor(ss, off, 32);
  const float inv = 1.0f / sqrtf(ss);

#pragma unroll
  for (int i = 0; i < 4; ++i) {
    ushort4 H, L;
    unsigned short h;
    float y;
    y = v[4*i+0] * inv; h = f2bf(y); H.x = h; L.x = f2bf(y - bf2f(h));
    y = v[4*i+1] * inv; h = f2bf(y); H.y = h; L.y = f2bf(y - bf2f(h));
    y = v[4*i+2] * inv; h = f2bf(y); H.z = h; L.z = f2bf(y - bf2f(h));
    y = v[4*i+3] * inv; h = f2bf(y); H.w = h; L.w = f2bf(y - bf2f(h));
    *(ushort4*)(Xhi + (size_t)row * DDIM + (lane + 32 * i) * 4) = H;
    *(ushort4*)(Xlo + (size_t)row * DDIM + (lane + 32 * i) * 4) = L;
  }
  if (lane == 0) {            // sentinels: minpos = +inf (u32 max), maxneg = -inf (u32 0)
    minpos[row] = 0xFFFFFFFFu;
    maxneg[row] = 0u;
  }
}

// ---------------------------------------------------------------------------
// Kernel 2: fused Gram tile + masked min/max.
// Block: 64x128 tile of G. 8 waves: wave = (colHalf<<2)|rowTile.
// Each wave: 16 rows x 64 cols = 4 accumulators (v8f). K streamed in KC=32
// chunks through double-buffered LDS via async DMA; 3 WMMAs per (tile,chunk).
// ---------------------------------------------------------------------------
__global__ __launch_bounds__(256) void gram_minmax_kernel(
    const unsigned short* __restrict__ Xhi, const unsigned short* __restrict__ Xlo,
    const int* __restrict__ targets,
    unsigned* __restrict__ minpos, unsigned* __restrict__ maxneg)
{
  __shared__ __align__(16) unsigned short slds[2 * BUFBYTES / 2];

  const int Rbase = blockIdx.x * BM;
  const int Cbase = blockIdx.y * BN;
  const int tid   = threadIdx.x;
  const int lane  = tid & 31;
  const int wave  = tid >> 5;
  const int rt    = wave & 3;    // row tile 0..3 (16 rows each)
  const int ch    = wave >> 2;   // column half 0..1 (4 col tiles each)
  const int m     = lane & 15;   // row/col index inside tile
  const int kh    = lane >> 4;   // K-half selector (ISA 16-bit A/B layout)

  // ---- per-thread DMA addressing (6 x b128 per chunk: Ahi,Alo,Bhi0,Blo0,Bhi1,Blo1) ----
  const unsigned ldsbase = (unsigned)(unsigned long long)(void*)slds; // low 32b = LDS offset
  const unsigned rowA = tid >> 2, q = tid & 3;            // A: 64 rows x 4 16B quarters
  const unsigned rowB1 = rowA + 64;                       // B: 128 rows, two halves
  const unsigned ldsA  = (rowA  * LDT + q * 8) * 2;
  const unsigned ldsB0 = (rowA  * LDT + q * 8) * 2;
  const unsigned ldsB1 = (rowB1 * LDT + q * 8) * 2;
  const unsigned voffA  = (((unsigned)(Rbase) + rowA ) * DDIM + q * 8) * 2;
  const unsigned voffB0 = (((unsigned)(Cbase) + rowA ) * DDIM + q * 8) * 2;
  const unsigned voffB1 = (((unsigned)(Cbase) + rowB1) * DDIM + q * 8) * 2;

  v8f acc[4] = {};

  // prologue: DMA chunk 0 into buffer 0
  {
    const unsigned bb = ldsbase;
    async_ld_b128(bb + OFF_AHI + ldsA,  Xhi, voffA);
    async_ld_b128(bb + OFF_ALO + ldsA,  Xlo, voffA);
    async_ld_b128(bb + OFF_BHI + ldsB0, Xhi, voffB0);
    async_ld_b128(bb + OFF_BLO + ldsB0, Xlo, voffB0);
    async_ld_b128(bb + OFF_BHI + ldsB1, Xhi, voffB1);
    async_ld_b128(bb + OFF_BLO + ldsB1, Xlo, voffB1);
  }

  for (int it = 0; it < NCHUNK; ++it) {
    const int cur = it & 1;
    if (it + 1 < NCHUNK) {
      // DMA next chunk into the other buffer (overlaps with this chunk's WMMAs)
      const unsigned kb = (unsigned)(it + 1) * (KC * 2);
      const unsigned bb = ldsbase + (unsigned)(cur ^ 1) * BUFBYTES;
      async_ld_b128(bb + OFF_AHI + ldsA,  Xhi, voffA  + kb);
      async_ld_b128(bb + OFF_ALO + ldsA,  Xlo, voffA  + kb);
      async_ld_b128(bb + OFF_BHI + ldsB0, Xhi, voffB0 + kb);
      async_ld_b128(bb + OFF_BLO + ldsB0, Xlo, voffB0 + kb);
      async_ld_b128(bb + OFF_BHI + ldsB1, Xhi, voffB1 + kb);
      async_ld_b128(bb + OFF_BLO + ldsB1, Xlo, voffB1 + kb);
      wait_async_le6();     // in-order completion => current chunk's 6 loads landed
    } else {
      wait_async_0();
    }
    __syncthreads();        // all waves' current-buffer DMA complete

    const unsigned short* pbuf = slds + (size_t)cur * (BUFBYTES / 2);
    const unsigned short* pAhi = pbuf;
    const unsigned short* pAlo = pbuf + OFF_ALO / 2;
    const unsigned short* pBhi = pbuf + OFF_BHI / 2;
    const unsigned short* pBlo = pbuf + OFF_BLO / 2;

    // ---- A fragments (this wave's 16-row tile), ISA 16-bit A layout:
    // lane m,kh: VGPRs 0..3 hold k = 8*kh+0..7 ; VGPRs 4..7 hold k = 16+8*kh+0..7
    Frag ahi, alo;
    {
      const int ar = rt * 16 + m;
      ahi.h[0] = *(const v8bf*)&pAhi[ar * LDT + 8 * kh];
      ahi.h[1] = *(const v8bf*)&pAhi[ar * LDT + 16 + 8 * kh];
      alo.h[0] = *(const v8bf*)&pAlo[ar * LDT + 8 * kh];
      alo.h[1] = *(const v8bf*)&pAlo[ar * LDT + 16 + 8 * kh];
    }

#pragma unroll
    for (int t = 0; t < 4; ++t) {
      const int br = (ch * 4 + t) * 16 + m;   // B(k,n): n-th column = row-major row of X
      Frag bhi, blo;
      bhi.h[0] = *(const v8bf*)&pBhi[br * LDT + 8 * kh];
      bhi.h[1] = *(const v8bf*)&pBhi[br * LDT + 16 + 8 * kh];
      blo.h[0] = *(const v8bf*)&pBlo[br * LDT + 8 * kh];
      blo.h[1] = *(const v8bf*)&pBlo[br * LDT + 16 + 8 * kh];
      acc[t] = __builtin_amdgcn_wmma_f32_16x16x32_bf16(false, ahi.v, false, bhi.v,
                                                       (short)0, acc[t], false, false);
      acc[t] = __builtin_amdgcn_wmma_f32_16x16x32_bf16(false, ahi.v, false, blo.v,
                                                       (short)0, acc[t], false, false);
      acc[t] = __builtin_amdgcn_wmma_f32_16x16x32_bf16(false, alo.v, false, bhi.v,
                                                       (short)0, acc[t], false, false);
    }
    __syncthreads();        // protect this buffer: iteration it+1 DMAs chunk it+2 into it
  }

  // ---- fused epilogue: masked min/max over this 64x128 tile ----
  // C/D layout: VGPR r holds row (r + 8*kh), lane m is the column within tile.
  int tr[8];
#pragma unroll
  for (int r = 0; r < 8; ++r) tr[r] = targets[Rbase + rt * 16 + r + 8 * kh];

  float mp[8], mn[8];
#pragma unroll
  for (int r = 0; r < 8; ++r) { mp[r] = FLTMAXF; mn[r] = -FLTMAXF; }

#pragma unroll
  for (int t = 0; t < 4; ++t) {
    const int tc = targets[Cbase + (ch * 4 + t) * 16 + m];
#pragma unroll
    for (int r = 0; r < 8; ++r) {
      const float s = acc[t][r];
      const bool same = (tr[r] == tc);
      if (same)  { if (s < mp[r]) mp[r] = s; }
      else       { if (s > mn[r]) mn[r] = s; }
    }
  }
  // reduce across the 16 lanes of each half-wave (xor 1,2,4,8 stays in-half)
#pragma unroll
  for (int off = 1; off <= 8; off <<= 1) {
#pragma unroll
    for (int r = 0; r < 8; ++r) {
      mp[r] = fminf(mp[r], __shfl_xor(mp[r], off, 32));
      mn[r] = fmaxf(mn[r], __shfl_xor(mn[r], off, 32));
    }
  }
  if (m == 0) {                // lanes 0 and 16 commit their 8 rows each
#pragma unroll
    for (int r = 0; r < 8; ++r) {
      const int row = Rbase + rt * 16 + r + 8 * kh;
      atomicMin(&minpos[row], encf(mp[r]));
      atomicMax(&maxneg[row], encf(mn[r]));
    }
  }
}

// ---------------------------------------------------------------------------
// Kernel 3: deterministic final reduction -> scalar loss.
// ---------------------------------------------------------------------------
__global__ __launch_bounds__(256) void loss_reduce_kernel(
    const unsigned* __restrict__ minpos, const unsigned* __restrict__ maxneg,
    float* __restrict__ out)
{
  __shared__ float sm[256];
  float acc = 0.f;
  for (int i = threadIdx.x; i < NROWS; i += 256) {
    const float dist_ap = -decf(minpos[i]);           // -(hardest positive similarity)
    const unsigned un = maxneg[i];
    if (un != 0u) {                                    // un==0 => no negatives => hinge 0
      const float dist_an = -decf(un);
      float t = dist_ap - dist_an + MARGINF;
      acc += (t > 0.f) ? t : 0.f;
    }
  }
  sm[threadIdx.x] = acc;
  __syncthreads();
  for (int s = 128; s > 0; s >>= 1) {
    if (threadIdx.x < s) sm[threadIdx.x] += sm[threadIdx.x + s];
    __syncthreads();
  }
  if (threadIdx.x == 0) out[0] = sm[0] / (float)NROWS;
}

// ---------------------------------------------------------------------------
extern "C" void kernel_launch(void* const* d_in, const int* in_sizes, int n_in,
                              void* d_out, int out_size, void* d_ws, size_t ws_size,
                              hipStream_t stream) {
  const float* X       = (const float*)d_in[0];
  const int*   targets = (const int*)d_in[1];

  // workspace layout: Xhi (8MB) | Xlo (8MB) | minpos (32KB) | maxneg (32KB)
  unsigned short* Xhi = (unsigned short*)d_ws;
  unsigned short* Xlo = Xhi + (size_t)NROWS * DDIM;
  unsigned* minpos    = (unsigned*)(Xlo + (size_t)NROWS * DDIM);
  unsigned* maxneg    = minpos + NROWS;

  norm_split_kernel<<<dim3(NROWS / 8), dim3(256), 0, stream>>>(X, Xhi, Xlo, minpos, maxneg);
  gram_minmax_kernel<<<dim3(NROWS / BM, NROWS / BN), dim3(256), 0, stream>>>(
      Xhi, Xlo, targets, minpos, maxneg);
  loss_reduce_kernel<<<dim3(1), dim3(256), 0, stream>>>(minpos, maxneg, (float*)d_out);
}
